// CausalSelfAttention_29841432773330
// MI455X (gfx1250) — matrix operand order repaired and checked by reading, not compiled, not run on hardware
//
#include <hip/hip_runtime.h>
#include <hip/hip_bf16.h>

// ---------- types ----------
typedef __attribute__((ext_vector_type(16))) __bf16 bf16x16;
typedef __attribute__((ext_vector_type(8)))  float  f32x8;
typedef __attribute__((ext_vector_type(4)))  unsigned int u32x4;
typedef __attribute__((ext_vector_type(8)))  int i32x8;
typedef __attribute__((ext_vector_type(4)))  int i32x4;

union FragA {
    bf16x16 v;
    uint4   q[2];
};

__device__ __forceinline__ unsigned short f2bf(float f) {
    unsigned u = __float_as_uint(f);
    unsigned r = u + 0x7FFFu + ((u >> 16) & 1u);   // round-to-nearest-even
    return (unsigned short)(r >> 16);
}

__device__ __forceinline__ f32x8 wmma_bf16(const FragA& a, const FragA& b, f32x8 c) {
    return __builtin_amdgcn_wmma_f32_16x16x32_bf16(
        /*neg_a=*/false, a.v, /*neg_b=*/false, b.v,
        /*c_mod=*/(short)0, c, /*reuse_a=*/false, /*reuse_b=*/false);
}

// Low 32 bits of a flat pointer into the LDS aperture == LDS byte offset
// (flat-to-LDS mapping, see the CDNA5 ISA).
__device__ __forceinline__ unsigned lds_lo(const void* p) {
    return (unsigned)(uintptr_t)p;
}

// ---------- TDM: 2-D bf16 tile load, global -> LDS ----------
// Loads tile_d1 rows x tile_d0 elements (2B each) starting at gptr, source row
// stride = stride_d0 elements. TDM padding inserts 4 DWORDs (8 ushorts) after
// every 16 DWORDs (32 ushorts) stored, producing an LDS row stride of 40
// ushorts (= LDSP below).
__device__ __forceinline__ void tdm_load_2d(
    unsigned lds_byte_addr, const void* gptr,
    unsigned tensor_d0, unsigned tensor_d1,
    unsigned tile_d0, unsigned tile_d1, unsigned stride_d0)
{
    unsigned long long ga = (unsigned long long)(uintptr_t)gptr;
    u32x4 g0;
    g0[0] = 1u;                                   // count=1, user mode
    g0[1] = lds_byte_addr;                        // lds_addr
    g0[2] = (unsigned)ga;                         // global_addr[31:0]
    g0[3] = (unsigned)(ga >> 32) | (2u << 30);    // global_addr[56:32] | type=2

    // group1: data_size=1 (2B), pad_enable=1, pad_interval=3 (16 DW),
    // pad_amount=3 (4 DW); no multicast, no atomic barrier, no iterate.
    unsigned w0 = (1u << 16) | (1u << 20) | (3u << 22) | (3u << 25);
    unsigned w1 = (tensor_d0 & 0xFFFFu) << 16;                    // abar=0 | dim0.lo
    unsigned w2 = (tensor_d0 >> 16) | ((tensor_d1 & 0xFFFFu) << 16);
    unsigned w3 = (tensor_d1 >> 16) | (tile_d0 << 16);
    unsigned w4 = tile_d1;                                        // tile_dim1 | tile_dim2=0
    unsigned w5 = stride_d0;                                      // dim0_stride[31:0]
    i32x8 g1 = { (int)w0, (int)w1, (int)w2, (int)w3,
                 (int)w4, (int)w5, 0, 0 };
    i32x4 z4 = { 0, 0, 0, 0 };
    i32x8 z8 = { 0, 0, 0, 0, 0, 0, 0, 0 };
    // 6-arg form on this toolchain: (g0, g1, g2, g3, ext, cpol)
    __builtin_amdgcn_tensor_load_to_lds(g0, g1, z4, z4, z8, 0);
}

// ---------- constants (problem shape is fixed) ----------
#define BB 4
#define LL 2048
#define DD 1024
#define HH 16
#define HD 64

// ---------- f32 -> bf16 convert ----------
__global__ __launch_bounds__(256) void cvt_f32_bf16(const float* __restrict__ in,
                                                    unsigned short* __restrict__ out, int n) {
    for (int i = blockIdx.x * blockDim.x + threadIdx.x; i < n; i += gridDim.x * blockDim.x)
        out[i] = f2bf(in[i]);
}

// ---------- GEMM: C[M,N] = A[M,K] @ W[N,K]^T + bias ----------
// mode 0: write f32 row-major [M,N] to outF
// mode 1: write bf16 scattered as [B,H,L,HD]   (Q,K)
// mode 2: write bf16 scattered as [B,H,HD,L]   (V transposed)
#define TM 128
#define TN 64
#define TK 32
#define LDSP (TK + 8)   // padded stride (ushorts) — must match TDM pad config

__global__ __launch_bounds__(256) void gemm_bf16_wmma(
    const unsigned short* __restrict__ A,
    const unsigned short* __restrict__ W,
    const float* __restrict__ bias,
    float* __restrict__ outF,
    unsigned short* __restrict__ outBF,
    int M, int N, int K, int mode)
{
    __shared__ __align__(16) unsigned short As[2][TM][LDSP];
    __shared__ __align__(16) unsigned short Bs[2][TN][LDSP];

    const int tid  = threadIdx.x;
    const int lane = tid & 31;
    const int wid  = tid >> 5;
    const int bm   = blockIdx.y * TM;
    const int bn   = blockIdx.x * TN;
    const int wm   = (wid >> 1) * 32;   // 0,32,64,96
    const int wn   = (wid & 1) * 32;    // 0,32

    const int m16   = lane & 15;
    const int ak    = (lane & 16) ? 16 : 0;  // A-frag K sub-offset
    const int bk    = (lane & 16) ? 16 : 0;  // B-frag K sub-offset
    const int half8 = (lane & 16) ? 8 : 0;

    f32x8 acc[2][2] = {};

    // prologue: wave 0 kicks off the DMA for the first K-slab
    if (wid == 0) {
        tdm_load_2d(lds_lo(&As[0][0][0]), A + (size_t)bm * K, K, M - bm, TK, TM, K);
        tdm_load_2d(lds_lo(&Bs[0][0][0]), W + (size_t)bn * K, K, N - bn, TK, TN, K);
    }

    int cur = 0;
    for (int k0 = 0; k0 < K; k0 += TK, cur ^= 1) {
        if (wid == 0) __builtin_amdgcn_s_wait_tensorcnt(0);  // cur slab landed in LDS
        __syncthreads();
        if (wid == 0 && k0 + TK < K) {                       // DMA next slab while computing
            tdm_load_2d(lds_lo(&As[cur ^ 1][0][0]),
                        A + (size_t)bm * K + (k0 + TK), K, M - bm, TK, TM, K);
            tdm_load_2d(lds_lo(&Bs[cur ^ 1][0][0]),
                        W + (size_t)bn * K + (k0 + TK), K, N - bn, TK, TN, K);
        }

        FragA a[2], b[2];
        for (int mi = 0; mi < 2; ++mi) {
            int r = wm + mi * 16 + m16;
            a[mi].q[0] = *(const uint4*)&As[cur][r][ak];
            a[mi].q[1] = *(const uint4*)&As[cur][r][ak + 8];
        }
        for (int ni = 0; ni < 2; ++ni) {
            int r = wn + ni * 16 + m16;
            b[ni].q[0] = *(const uint4*)&Bs[cur][r][bk];
            b[ni].q[1] = *(const uint4*)&Bs[cur][r][bk + 8];
        }
        for (int mi = 0; mi < 2; ++mi)
            for (int ni = 0; ni < 2; ++ni)
                acc[mi][ni] = wmma_bf16(a[mi], b[ni], acc[mi][ni]);
    }

    // epilogue
    for (int mi = 0; mi < 2; ++mi) {
        for (int ni = 0; ni < 2; ++ni) {
            const int col = bn + wn + ni * 16 + m16;
            const float bv = bias ? bias[col] : 0.f;
            for (int v = 0; v < 8; ++v) {
                const int row = bm + wm + mi * 16 + v + half8;
                const float val = acc[mi][ni][v] + bv;
                if (mode == 0) {
                    outF[(size_t)row * N + col] = val;
                } else {
                    const int b  = row >> 11, l  = row & (LL - 1);
                    const int h  = col >> 6,  hd = col & (HD - 1);
                    size_t idx = (mode == 1)
                        ? (((size_t)(b * HH + h) * LL + l) * HD + hd)
                        : (((size_t)(b * HH + h) * HD + hd) * LL + l);
                    outBF[idx] = f2bf(val);
                }
            }
        }
    }
}

// ---------- flash attention ----------
// Q,K: [B,H,L,HD] bf16;  Vt: [B,H,HD,L] bf16;  O: [B*L, D] bf16
__global__ __launch_bounds__(128) void flash_attn(
    const unsigned short* __restrict__ Q,
    const unsigned short* __restrict__ Kt,
    const unsigned short* __restrict__ Vt,
    const int* __restrict__ mask,
    unsigned short* __restrict__ O)
{
    __shared__ __align__(16) unsigned short Pld[4][16 * 32];

    const int lane = threadIdx.x & 31;
    const int w    = threadIdx.x >> 5;
    const int bh   = blockIdx.y;           // b*H + h
    const int b    = bh >> 4;
    const int h    = bh & (HH - 1);
    const int q0   = blockIdx.x * 64 + w * 16;

    const unsigned short* Qh = Q  + (size_t)bh * LL * HD;
    const unsigned short* Kh = Kt + (size_t)bh * LL * HD;
    const unsigned short* Vh = Vt + (size_t)bh * HD * LL;
    const int* mb = mask + b * LL;

    const int m16   = lane & 15;
    const int ak    = (lane & 16) ? 16 : 0;
    const int bk    = (lane & 16) ? 16 : 0;
    const int half8 = (lane & 16) ? 8 : 0;

    // Q fragments (persistent): head dim 64 = 2 x K32 A-frags
    FragA aq[2];
    for (int kk = 0; kk < 2; ++kk) {
        const unsigned short* qrow = Qh + (size_t)(q0 + m16) * HD + kk * 32;
        aq[kk].q[0] = *(const uint4*)(qrow + ak);
        aq[kk].q[1] = *(const uint4*)(qrow + ak + 8);
    }

    f32x8 acc[4] = {};
    float mrow[8], lrow[8];
    for (int v = 0; v < 8; ++v) { mrow[v] = -1e30f; lrow[v] = 0.f; }

    const float scale = 0.125f;   // HD^-0.5
    const int kend = q0 + 16;     // causal: keys <= q0+15

    for (int k0 = 0; k0 < kend; k0 += 32) {
        // ---- S = Q K^T, two 16x16 tiles ----
        f32x8 s[2];
        for (int sub = 0; sub < 2; ++sub) {
            const unsigned short* krow = Kh + (size_t)(k0 + sub * 16 + m16) * HD;
            FragA bk0, bk1;
            bk0.q[0] = *(const uint4*)(krow + bk);
            bk0.q[1] = *(const uint4*)(krow + bk + 8);
            bk1.q[0] = *(const uint4*)(krow + 32 + bk);
            bk1.q[1] = *(const uint4*)(krow + 32 + bk + 8);
            f32x8 c = {};
            c = wmma_bf16(aq[0], bk0, c);
            c = wmma_bf16(aq[1], bk1, c);
            s[sub] = c;
        }

        // ---- mask + online softmax (row stats replicated across 16-lane halves) ----
        const int kg0 = k0 + m16;
        const int pm0 = mb[kg0];
        const int pm1 = mb[kg0 + 16];
        float pscale[8];
        for (int v = 0; v < 8; ++v) {
            const int qg = q0 + v + half8;
            float s0 = s[0][v] * scale; if (kg0 > qg      || pm0 == 0) s0 = -1e9f;
            float s1 = s[1][v] * scale; if (kg0 + 16 > qg || pm1 == 0) s1 = -1e9f;
            float r = fmaxf(s0, s1);
            for (int off = 8; off; off >>= 1) r = fmaxf(r, __shfl_xor(r, off, 32));
            const float mnew = fmaxf(mrow[v], r);
            const float sc   = __expf(mrow[v] - mnew);
            mrow[v] = mnew;
            const float p0 = __expf(s0 - mnew);
            const float p1 = __expf(s1 - mnew);
            float rs = p0 + p1;
            for (int off = 8; off; off >>= 1) rs += __shfl_xor(rs, off, 32);
            lrow[v] = lrow[v] * sc + rs;
            pscale[v] = sc;
            s[0][v] = p0; s[1][v] = p1;
        }
        for (int ni = 0; ni < 4; ++ni)
            for (int v = 0; v < 8; ++v)
                acc[ni][v] *= pscale[v];

        // ---- relayout P (C-frag) -> A-frag via per-wave LDS tile ----
        unsigned short* P = &Pld[w][0];
        for (int v = 0; v < 8; ++v) {
            const int r = v + half8;
            P[r * 32 + m16]      = f2bf(s[0][v]);
            P[r * 32 + 16 + m16] = f2bf(s[1][v]);
        }
        __builtin_amdgcn_wave_barrier();
        FragA ap;
        ap.q[0] = *(const uint4*)&P[m16 * 32 + ak];
        ap.q[1] = *(const uint4*)&P[m16 * 32 + ak + 8];

        // ---- acc += P @ V  (V stored transposed -> contiguous B-frags) ----
        for (int ni = 0; ni < 4; ++ni) {
            const unsigned short* vrow = Vh + (size_t)(ni * 16 + m16) * LL + k0;
            FragA bv;
            bv.q[0] = *(const uint4*)(vrow + bk);
            bv.q[1] = *(const uint4*)(vrow + bk + 8);
            acc[ni] = wmma_bf16(ap, bv, acc[ni]);
        }
    }

    // ---- normalize + store bf16 [B*L, D] ----
    for (int v = 0; v < 8; ++v) {
        const float inv = 1.f / lrow[v];
        const int qg = q0 + v + half8;
        const size_t rbase = ((size_t)b * LL + qg) * DD + h * HD;
        for (int ni = 0; ni < 4; ++ni)
            O[rbase + ni * 16 + m16] = f2bf(acc[ni][v] * inv);
    }
}

// ---------- host side ----------
extern "C" void kernel_launch(void* const* d_in, const int* in_sizes, int n_in,
                              void* d_out, int out_size, void* d_ws, size_t ws_size,
                              hipStream_t stream) {
    (void)in_sizes; (void)n_in; (void)out_size; (void)ws_size;

    const float* x   = (const float*)d_in[0];
    const int*   msk = (const int*)d_in[1];
    const float* Wq  = (const float*)d_in[2];
    const float* bq  = (const float*)d_in[3];
    const float* Wk  = (const float*)d_in[4];
    const float* bk_ = (const float*)d_in[5];
    const float* Wv  = (const float*)d_in[6];
    const float* bv_ = (const float*)d_in[7];
    const float* Wo  = (const float*)d_in[8];
    const float* bo  = (const float*)d_in[9];
    float* out = (float*)d_out;

    const size_t M   = (size_t)BB * LL;                // 8192
    const size_t SZX = M * DD * 2;                     // 16 MB bf16
    const size_t SZW = (size_t)DD * DD * 2;            // 2 MB bf16
    const size_t SZH = (size_t)BB * HH * LL * HD * 2;  // 16 MB bf16

    char* ws = (char*)d_ws;
    unsigned short* xbf  = (unsigned short*)(ws);
    unsigned short* wqb  = (unsigned short*)(ws + SZX);
    unsigned short* wkb  = (unsigned short*)(ws + SZX + 1 * SZW);
    unsigned short* wvb  = (unsigned short*)(ws + SZX + 2 * SZW);
    unsigned short* wob  = (unsigned short*)(ws + SZX + 3 * SZW);
    unsigned short* Qb   = (unsigned short*)(ws + SZX + 4 * SZW);
    unsigned short* Kb   = (unsigned short*)(ws + SZX + 4 * SZW + 1 * SZH);
    unsigned short* Vtb  = (unsigned short*)(ws + SZX + 4 * SZW + 2 * SZH);
    unsigned short* attO = (unsigned short*)(ws + SZX + 4 * SZW + 3 * SZH);

    // 1) bf16 conversions
    cvt_f32_bf16<<<4096, 256, 0, stream>>>(x,  xbf, (int)(M * DD));
    cvt_f32_bf16<<<1024, 256, 0, stream>>>(Wq, wqb, DD * DD);
    cvt_f32_bf16<<<1024, 256, 0, stream>>>(Wk, wkb, DD * DD);
    cvt_f32_bf16<<<1024, 256, 0, stream>>>(Wv, wvb, DD * DD);
    cvt_f32_bf16<<<1024, 256, 0, stream>>>(Wo, wob, DD * DD);

    // 2) projections (Q,K -> [B,H,L,64]; V -> [B,H,64,L])
    dim3 ggrid(DD / TN, (unsigned)(M / TM));
    gemm_bf16_wmma<<<ggrid, 256, 0, stream>>>(xbf, wqb, bq,  nullptr, Qb,  (int)M, DD, DD, 1);
    gemm_bf16_wmma<<<ggrid, 256, 0, stream>>>(xbf, wkb, bk_, nullptr, Kb,  (int)M, DD, DD, 1);
    gemm_bf16_wmma<<<ggrid, 256, 0, stream>>>(xbf, wvb, bv_, nullptr, Vtb, (int)M, DD, DD, 2);

    // 3) causal flash attention
    flash_attn<<<dim3(LL / 64, BB * HH), 128, 0, stream>>>(Qb, Kb, Vtb, msk, attO);

    // 4) output projection (f32 out + bias)
    gemm_bf16_wmma<<<ggrid, 256, 0, stream>>>(attO, wob, bo, out, nullptr, (int)M, DD, DD, 0);
}
